// StableGenerator_19636590478031
// MI455X (gfx1250) — matrix operand order repaired and checked by reading, not compiled, not run on hardware
//
#include <hip/hip_runtime.h>
#include <hip/hip_bf16.h>

// ---------------------------------------------------------------------------
// out[2M,64] = z[2M,64] @ expm(G*dt),  G = -(A^T A) + (B - B^T)
//
// Kernel 1: one workgroup computes Delta = expm(G*dt) - I in LDS -> d_ws.
// Kernel 2: out = z + (z_f16 @ Delta_f16) via v_wmma_f32_16x16x32_f16.
//           Near-identity split keeps fp32-level accuracy while using the
//           f16 WMMA pipe; kernel is HBM-bandwidth bound (~1 GB @ 23.3 TB/s).
// ---------------------------------------------------------------------------

typedef __attribute__((ext_vector_type(16))) _Float16 v16h;
typedef __attribute__((ext_vector_type(8)))  float    v8f;
typedef __attribute__((ext_vector_type(4)))  float    f32x4;

#define DD 64
#define ROW_PAD 68          // 16B-aligned row pitch, rotates LDS banks by 4/row
#define WAVES_PER_BLOCK 8

// ------------------------- Kernel 1: matrix exponential --------------------
__global__ __launch_bounds__(1024) void expm_delta_kernel(
    const float* __restrict__ A, const float* __restrict__ B,
    const int* __restrict__ dtp, float* __restrict__ delta) {
  __shared__ float sT[DD * DD];   // term   (stages A first)
  __shared__ float sC[DD * DD];   // acc    (stages B first)
  __shared__ float sM[DD * DD];   // Ms = G*dt / 2^6
  const int tid = threadIdx.x;
  const int j  = tid & 63;
  const int i0 = tid >> 6;        // [0,16); each thread owns rows i0+16t, col j

#pragma unroll
  for (int t = 0; t < 4; ++t) {
    sT[tid + 1024 * t] = A[tid + 1024 * t];
    sC[tid + 1024 * t] = B[tid + 1024 * t];
  }
  __syncthreads();

  const float scale = (float)dtp[0] * (1.0f / 64.0f);
#pragma unroll
  for (int t = 0; t < 4; ++t) {
    const int i = i0 + 16 * t;
    float g = sC[i * DD + j] - sC[j * DD + i];
    for (int k = 0; k < DD; ++k)
      g -= sT[k * DD + i] * sT[k * DD + j];    // -(A^T A)
    sM[i * DD + j] = g * scale;
  }
  __syncthreads();

  // term = I, acc = I
#pragma unroll
  for (int t = 0; t < 4; ++t) {
    const int i = i0 + 16 * t;
    const float id = (i == j) ? 1.0f : 0.0f;
    sT[i * DD + j] = id;
    sC[i * DD + j] = id;
  }
  __syncthreads();

  // Taylor: term = term @ Ms / k ; acc += term   (k = 1..12)
  for (int k = 1; k <= 12; ++k) {
    float r[4];
    const float invk = 1.0f / (float)k;
#pragma unroll
    for (int t = 0; t < 4; ++t) {
      const int i = i0 + 16 * t;
      float acc = 0.0f;
      for (int m = 0; m < DD; ++m)
        acc += sT[i * DD + m] * sM[m * DD + j];
      r[t] = acc * invk;
    }
    __syncthreads();
#pragma unroll
    for (int t = 0; t < 4; ++t) {
      const int i = i0 + 16 * t;
      sT[i * DD + j] = r[t];
      sC[i * DD + j] += r[t];
    }
    __syncthreads();
  }

  // 6 squarings; on the last one write Delta = acc - I straight from regs
  for (int s = 0; s < 6; ++s) {
    float r[4];
#pragma unroll
    for (int t = 0; t < 4; ++t) {
      const int i = i0 + 16 * t;
      float acc = 0.0f;
      for (int m = 0; m < DD; ++m)
        acc += sC[i * DD + m] * sC[m * DD + j];
      r[t] = acc;
    }
    __syncthreads();
    if (s < 5) {
#pragma unroll
      for (int t = 0; t < 4; ++t)
        sC[(i0 + 16 * t) * DD + j] = r[t];
      __syncthreads();
    } else {
#pragma unroll
      for (int t = 0; t < 4; ++t) {
        const int i = i0 + 16 * t;
        delta[i * DD + j] = r[t] - ((i == j) ? 1.0f : 0.0f);
      }
    }
  }
}

// ------------------------- Kernel 2: z @ m via WMMA ------------------------
__global__ __launch_bounds__(256) void zmatmul_kernel(
    const float* __restrict__ z, const float* __restrict__ delta,
    float* __restrict__ out, int batch) {
  __shared__ float lds[WAVES_PER_BLOCK * 16 * ROW_PAD];
  const int lane   = threadIdx.x & 31;
  const int wave   = threadIdx.x >> 5;
  const int colN   = lane & 15;
  const int hiHalf = lane >> 4;            // 0: lanes 0-15, 1: lanes 16-31
  float* zs = lds + wave * 16 * ROW_PAD;

  // B fragments of Delta (32K x 16N tiles), per ISA 16-bit B layout:
  // lane n (0-15): K rows 0..7,16..23 of the 32-K slab; lane n+16: K rows 8..15,24..31.
  v16h bfrag[2][4];
  {
    const int kb = hiHalf * 8;
#pragma unroll
    for (int h = 0; h < 2; ++h)
#pragma unroll
      for (int n = 0; n < 4; ++n) {
        v16h b;
#pragma unroll
        for (int jj = 0; jj < 16; ++jj)
          b[jj] = (_Float16)delta[(h * 32 + kb + (jj & 7) + (jj >> 3) * 16) * DD + n * 16 + colN];
        bfrag[h][n] = b;
      }
  }

  const int M0 = hiHalf * 8;
  const int kb = hiHalf * 8;
  const int nstrips = (batch + 15) >> 4;
  const long long waveStride = (long long)gridDim.x * WAVES_PER_BLOCK;

  for (long long strip = (long long)blockIdx.x * WAVES_PER_BLOCK + wave;
       strip < nstrips; strip += waveStride) {
    const long long r0 = strip << 4;
    const bool full = (r0 + 16) <= (long long)batch;   // wave-uniform

    // ---- Stage 16 rows x 64 cols of z into LDS ----
    if (full) {
      const f32x4* zp = (const f32x4*)(z + r0 * DD);   // one 64-bit base
#pragma unroll
      for (int i = 0; i < 8; ++i) {
        const int idx4 = lane + i * 32;                // [0,256)
        const int row  = idx4 >> 4;
        const int c4   = idx4 & 15;
        f32x4 v = __builtin_nontemporal_load(zp + idx4);
        *(f32x4*)(zs + row * ROW_PAD + c4 * 4) = v;
      }
    } else {
#pragma unroll
      for (int i = 0; i < 8; ++i) {
        const int idx4 = lane + i * 32;
        const int row  = idx4 >> 4;
        const int c4   = idx4 & 15;
        long long grow = r0 + row;
        if (grow >= batch) grow = batch - 1;           // clamp (tail strip)
        f32x4 v = *(const f32x4*)(z + grow * DD + c4 * 4);
        *(f32x4*)(zs + row * ROW_PAD + c4 * 4) = v;
      }
    }
    asm volatile("s_wait_dscnt 0" ::: "memory");       // cross-lane LDS RAW

    // ---- A fragments, ISA 16-bit A 16x32 layout ----
    // lane<16: halves 0-7 = K 0..7, halves 8-15 = K 16..23 (row = lane)
    // lane>=16: K 8..15 and K 24..31 (row = lane-16)
    const float* zr = zs + colN * ROW_PAD;
    v16h a0, a1;
#pragma unroll
    for (int jj = 0; jj < 8; ++jj) {
      a0[jj]     = (_Float16)zr[kb + jj];
      a0[jj + 8] = (_Float16)zr[kb + 16 + jj];
      a1[jj]     = (_Float16)zr[32 + kb + jj];
      a1[jj + 8] = (_Float16)zr[32 + kb + 16 + jj];
    }

    // ---- 8 back-to-back WMMAs: 4 independent N-tiles x 2 K-halves ----
    v8f cacc[4];
#pragma unroll
    for (int n = 0; n < 4; ++n) {
      v8f c = {};
      c = __builtin_amdgcn_wmma_f32_16x16x32_f16(false, a0, false, bfrag[0][n],
                                                 (short)0, c, false, false);
      c = __builtin_amdgcn_wmma_f32_16x16x32_f16(false, a1, false, bfrag[1][n],
                                                 (short)0, c, false, false);
      cacc[n] = c;
    }

    // ---- out = z + correction; C/D layout: c[j] -> (M = j + 8*hiHalf, N) ----
    const float* zadd = zs + M0 * ROW_PAD + colN;      // + const offsets below
    if (full) {
      float* op = out + r0 * DD + (M0 * DD + colN);    // one 64-bit base
#pragma unroll
      for (int n = 0; n < 4; ++n)
#pragma unroll
        for (int jj = 0; jj < 8; ++jj) {
          const float v = zadd[jj * ROW_PAD + n * 16] + cacc[n][jj];
          __builtin_nontemporal_store(v, op + jj * DD + n * 16);
        }
    } else {
#pragma unroll
      for (int n = 0; n < 4; ++n)
#pragma unroll
        for (int jj = 0; jj < 8; ++jj) {
          const long long row = r0 + M0 + jj;
          if (row < batch) {
            const float v = zadd[jj * ROW_PAD + n * 16] + cacc[n][jj];
            __builtin_nontemporal_store(v, out + row * DD + n * 16 + colN);
          }
        }
    }
  }
}

// ---------------------------------------------------------------------------
extern "C" void kernel_launch(void* const* d_in, const int* in_sizes, int n_in,
                              void* d_out, int out_size, void* d_ws, size_t ws_size,
                              hipStream_t stream) {
  const float* z  = (const float*)d_in[0];
  const float* A  = (const float*)d_in[1];
  const float* B  = (const float*)d_in[2];
  const int*   dt = (const int*)d_in[3];
  float* out   = (float*)d_out;
  float* delta = (float*)d_ws;               // 64*64 f32 = 16 KB scratch

  const int batch = in_sizes[0] / DD;

  expm_delta_kernel<<<1, 1024, 0, stream>>>(A, B, dt, delta);

  const int nstrips = (batch + 15) / 16;
  int blocks = (nstrips + WAVES_PER_BLOCK - 1) / WAVES_PER_BLOCK;
  if (blocks > 2048) blocks = 2048;
  zmatmul_kernel<<<blocks, 256, 0, stream>>>(z, delta, out, batch);
}
